// MultiHeadAttention_7636451852888
// MI455X (gfx1250) — matrix operand order repaired and checked by reading, not compiled, not run on hardware
//
#include <hip/hip_runtime.h>

// ---------------------------------------------------------------------------
// MultiHeadAttention forward for MI455X (gfx1250, wave32, WMMA 16x16x32 bf16)
// B=4, T=2048, DIM=768, HEADS=12, HEAD_DIM=64
// GEMMs: block = 256 threads (8 waves, 2Mx4N), block tile 32x256, per-wave
// C tile 16x64. Full 32x768 A strip DMA'd into LDS by the Tensor Data Mover
// (tensor_load_to_lds, TENSORcnt) -> barrier-free, double-buffered K loop.
// ---------------------------------------------------------------------------

#define DIMX   768
#define NQKV   2304
#define HEADS  12
#define HD     64
#define TSEQ   2048
#define BATCH  4
#define MROWS  (BATCH * TSEQ)   // 8192

typedef __bf16 bf16;
typedef __attribute__((ext_vector_type(16))) __bf16 v16bf;
typedef __attribute__((ext_vector_type(8)))  __bf16 v8bf;
typedef __attribute__((ext_vector_type(8)))  float  v8f;
typedef __attribute__((ext_vector_type(4)))  unsigned int u32x4;
typedef __attribute__((ext_vector_type(8)))  int i32x8;
typedef __attribute__((ext_vector_type(4)))  int i32x4;

union AF { v16bf v; v8bf h[2]; };

static __device__ __forceinline__ v8f wmma_bf16(const AF& a, const AF& b, v8f c) {
    return __builtin_amdgcn_wmma_f32_16x16x32_bf16(
        /*neg_a=*/false, a.v, /*neg_b=*/false, b.v,
        /*c_mod=*/(short)0, c, /*reuse_a=*/false, /*reuse_b=*/false);
}

// --------------------------- cast / transpose ------------------------------

__global__ void cast_bf16_kernel(const float* __restrict__ in, bf16* __restrict__ out, int n) {
    int i = blockIdx.x * blockDim.x + threadIdx.x;
    if (i < n) out[i] = (bf16)in[i];
}

// in[K][N] f32 -> out[N][K] bf16
__global__ void transpose_cast_kernel(const float* __restrict__ in, bf16* __restrict__ out,
                                      int K, int N) {
    int i = blockIdx.x * blockDim.x + threadIdx.x;
    if (i < K * N) {
        int k = i / N, n = i % N;
        out[(size_t)n * K + k] = (bf16)in[i];
    }
}

// ------------------- TDM: DMA a rows x 768 bf16 strip to LDS ---------------
// D# per CDNA5 ISA: group0 = {count, lds_addr,
// global_addr, type=2}; group1 = {data_size=2B, tensor_dim0/1, tile_dim0/1,
// tensor_dim0_stride}. 2D tile -> groups 2/3 zero. Issued by one wave only;
// EXEC is ignored by tensor ops. Tracked with TENSORcnt.
static __device__ __forceinline__ void tdm_load_strip(bf16* Als, const bf16* gsrc, int rows) {
    const unsigned long long ga = (unsigned long long)(size_t)gsrc;   // byte addr
    const unsigned int lds_off  = (unsigned int)(size_t)Als;          // LDS aperture low bits

    u32x4 g0;
    g0[0] = 1u;                                            // count=1 (valid user D#)
    g0[1] = lds_off;                                       // lds_addr [63:32]
    g0[2] = (unsigned int)(ga & 0xFFFFFFFFull);            // global_addr[31:0]
    g0[3] = (unsigned int)((ga >> 32) & 0x1FFFFFFull)      // global_addr[56:32]
          | 0x80000000u;                                   // type=2 ("image") [127:126]

    i32x8 g1;
    g1[0] = 1 << 16;                                       // data_size=1 -> 2 bytes
    g1[1] = (DIMX & 0xFFFF) << 16;                         // tensor_dim0[15:0] @ [63:48]
    g1[2] = (DIMX >> 16) | ((MROWS & 0xFFFF) << 16);       // dim0 hi | tensor_dim1 lo
    g1[3] = (MROWS >> 16) | (DIMX << 16);                  // dim1 hi | tile_dim0=768
    g1[4] = rows & 0xFFFF;                                 // tile_dim1=rows, tile_dim2=0
    g1[5] = DIMX;                                          // tensor_dim0_stride lo (elems)
    g1[6] = 0;                                             // stride0 hi | dim1_stride lo
    g1[7] = 0;

    const i32x4 gz = {0, 0, 0, 0};
#if __clang_major__ >= 23
    const i32x8 gz8 = {0, 0, 0, 0, 0, 0, 0, 0};
    __builtin_amdgcn_tensor_load_to_lds(g0, g1, gz, gz, gz8, 0);
#else
    __builtin_amdgcn_tensor_load_to_lds(g0, g1, gz, gz, 0);
#endif
}

// ----------------------- shared GEMM core ----------------------------------
// TDM-stages the 32x768 A strip (48KB LDS), then a barrier-free K loop with
// double-buffered B fragments: 4 accumulators = 16x64 C tile per wave.

struct GemmCore {
    v8f acc[4];
    int lr, half, wm, wn;
};

static __device__ __forceinline__ void gemm_core(
        GemmCore& g, bf16* Als,
        const bf16* __restrict__ X, const bf16* __restrict__ WT,
        int mblock, int nblock) {
    const int tid  = threadIdx.x;
    const int wave = tid >> 5;
    const int lane = tid & 31;
    g.lr   = lane & 15;
    g.half = lane >> 4;
    g.wm   = wave >> 2;     // 0..1
    g.wn   = wave & 3;      // 0..3

    // ---- A strip via Tensor Data Mover (wave 0 issues, all waves rendezvous)
    if (tid < 32) {
        tdm_load_strip(Als, X + (size_t)mblock * DIMX, 32);
        __builtin_amdgcn_s_wait_tensorcnt(0);
    }
    __syncthreads();

    // ---- barrier-free K loop, double-buffered B ----
    const int arow = g.wm * 16 + g.lr;
    const bf16* wrow[4];
    wrow[0] = WT + (size_t)(nblock + g.wn * 64 +  0 + g.lr) * DIMX;
    wrow[1] = WT + (size_t)(nblock + g.wn * 64 + 16 + g.lr) * DIMX;
    wrow[2] = WT + (size_t)(nblock + g.wn * 64 + 32 + g.lr) * DIMX;
    wrow[3] = WT + (size_t)(nblock + g.wn * 64 + 48 + g.lr) * DIMX;

    g.acc[0] = (v8f){}; g.acc[1] = (v8f){}; g.acc[2] = (v8f){}; g.acc[3] = (v8f){};

    auto loadB = [&](AF* bb, int kt) {
        #pragma unroll
        for (int j = 0; j < 4; ++j) {
            bb[j].h[0] = *(const v8bf*)(wrow[j] + kt + g.half * 16);
            bb[j].h[1] = *(const v8bf*)(wrow[j] + kt + g.half * 16 + 8);
        }
    };
    auto loadA = [&](int kt) {
        AF a;
        a.h[0] = *(const v8bf*)(&Als[arow * DIMX + kt + g.half * 16]);
        a.h[1] = *(const v8bf*)(&Als[arow * DIMX + kt + g.half * 16 + 8]);
        return a;
    };

    AF bc[4], bn[4];
    loadB(bc, 0);
    #pragma unroll 2
    for (int kt = 0; kt < DIMX - 32; kt += 32) {
        loadB(bn, kt + 32);                 // prefetch next K-step
        const AF a = loadA(kt);
        g.acc[0] = wmma_bf16(a, bc[0], g.acc[0]);
        g.acc[1] = wmma_bf16(a, bc[1], g.acc[1]);
        g.acc[2] = wmma_bf16(a, bc[2], g.acc[2]);
        g.acc[3] = wmma_bf16(a, bc[3], g.acc[3]);
        bc[0] = bn[0]; bc[1] = bn[1]; bc[2] = bn[2]; bc[3] = bn[3];
    }
    {
        const AF a = loadA(DIMX - 32);
        g.acc[0] = wmma_bf16(a, bc[0], g.acc[0]);
        g.acc[1] = wmma_bf16(a, bc[1], g.acc[1]);
        g.acc[2] = wmma_bf16(a, bc[2], g.acc[2]);
        g.acc[3] = wmma_bf16(a, bc[3], g.acc[3]);
    }
}

// ------------------------------ QKV GEMM -----------------------------------
// C[8192,2304] = Xbf16 @ Wqkv (W transposed [2304,768]). Epilogue scatters
// column c = d*36 + which*12 + h into Q/K (b,h,t,d) and V^T (b,h,d,t), bf16.
__global__ __launch_bounds__(256)
void qkv_gemm_kernel(const bf16* __restrict__ X, const bf16* __restrict__ WT,
                     bf16* __restrict__ Q, bf16* __restrict__ K, bf16* __restrict__ VT) {
    __shared__ __align__(16) bf16 Als[32 * DIMX];
    GemmCore g;
    const int mblock = blockIdx.y * 32;
    const int nblock = blockIdx.x * 256;
    gemm_core(g, Als, X, WT, mblock, nblock);

    #pragma unroll
    for (int j = 0; j < 4; ++j) {
        const int cidx  = nblock + g.wn * 64 + j * 16 + g.lr;  // 0..2303
        const int d     = cidx / 36;
        const int rem   = cidx % 36;
        const int which = rem / 12;
        const int h     = rem % 12;
        #pragma unroll
        for (int r = 0; r < 8; ++r) {
            const int m  = mblock + g.wm * 16 + r + 8 * g.half;
            const int bb = m >> 11;           // / 2048
            const int t  = m & 2047;
            const size_t bh = (size_t)(bb * HEADS + h);
            const bf16 val = (bf16)g.acc[j][r];
            if (which == 0)      Q [(bh * TSEQ + t) * HD + d] = val;
            else if (which == 1) K [(bh * TSEQ + t) * HD + d] = val;
            else                 VT[(bh * HD + d) * TSEQ + t] = val;
        }
    }
}

// ------------------------------ output GEMM --------------------------------
__global__ __launch_bounds__(256)
void out_gemm_kernel(const bf16* __restrict__ X, const bf16* __restrict__ WT,
                     const float* __restrict__ bias, float* __restrict__ OUT) {
    __shared__ __align__(16) bf16 Als[32 * DIMX];
    GemmCore g;
    const int mblock = blockIdx.y * 32;
    const int nblock = blockIdx.x * 256;
    gemm_core(g, Als, X, WT, mblock, nblock);

    #pragma unroll
    for (int j = 0; j < 4; ++j) {
        const int n = nblock + g.wn * 64 + j * 16 + g.lr;
        const float bv = bias[n];
        #pragma unroll
        for (int r = 0; r < 8; ++r) {
            const int m = mblock + g.wm * 16 + r + 8 * g.half;
            OUT[(size_t)m * DIMX + n] = g.acc[j][r] + bv;
        }
    }
}

// ---------------------------- flash attention ------------------------------
// One wave per (b, h, 16 query rows). Online softmax over 64 key-pair tiles
// (32 keys each): 2 WMMA for S = Q·K^T (contraction d=64), LDS round-trip to
// relayout P into an A fragment, then 4 WMMA for the 16x64 accumulator.
__global__ __launch_bounds__(32)
void attn_kernel(const bf16* __restrict__ Q, const bf16* __restrict__ Kt,
                 const bf16* __restrict__ VT, bf16* __restrict__ CTX) {
    __shared__ __align__(16) bf16 Pls[16 * 32];
    const int lane = threadIdx.x;
    const int lr   = lane & 15;
    const int half = lane >> 4;
    const int qt = blockIdx.x;   // 0..127
    const int h  = blockIdx.y;
    const int b  = blockIdx.z;
    const size_t bh = (size_t)(b * HEADS + h);

    const bf16* Qp = Q  + bh * TSEQ * HD;
    const bf16* Kp = Kt + bh * TSEQ * HD;
    const bf16* Vp = VT + bh * HD * TSEQ;

    // Q A-fragments for the two 32-wide contraction steps over d=64
    AF aq0, aq1;
    {
        const bf16* qrow = Qp + (size_t)(qt * 16 + lr) * HD;
        aq0.h[0] = *(const v8bf*)(qrow + half * 16);
        aq0.h[1] = *(const v8bf*)(qrow + half * 16 + 8);
        aq1.h[0] = *(const v8bf*)(qrow + 32 + half * 16);
        aq1.h[1] = *(const v8bf*)(qrow + 32 + half * 16 + 8);
    }

    float mrow[8], lrow[8], alpha[8];
    v8f acc[4] = {};
    #pragma unroll
    for (int r = 0; r < 8; ++r) { mrow[r] = -1e30f; lrow[r] = 0.0f; }

    for (int jp = 0; jp < TSEQ / 32; ++jp) {
        // S tiles for two 16-key blocks
        v8f s[2];
        #pragma unroll
        for (int u = 0; u < 2; ++u) {
            const int kb = jp * 32 + u * 16;
            const bf16* krow = Kp + (size_t)(kb + lr) * HD;
            AF b0, b1;
            b0.h[0] = *(const v8bf*)(krow + half * 16);
            b0.h[1] = *(const v8bf*)(krow + half * 16 + 8);
            b1.h[0] = *(const v8bf*)(krow + 32 + half * 16);
            b1.h[1] = *(const v8bf*)(krow + 32 + half * 16 + 8);
            v8f z = {};
            z = wmma_bf16(aq0, b0, z);
            z = wmma_bf16(aq1, b1, z);
            s[u] = z;
        }

        // online softmax (rows on VGPR index; cols across 16-lane group)
        #pragma unroll
        for (int r = 0; r < 8; ++r) {
            float v0 = s[0][r] * 0.125f;     // 1/sqrt(64)
            float v1 = s[1][r] * 0.125f;
            float mx = fmaxf(v0, v1);
            #pragma unroll
            for (int off = 1; off < 16; off <<= 1)
                mx = fmaxf(mx, __shfl_xor(mx, off, 32));
            const float mn = fmaxf(mrow[r], mx);
            const float p0 = __expf(v0 - mn);
            const float p1 = __expf(v1 - mn);
            float rs = p0 + p1;
            #pragma unroll
            for (int off = 1; off < 16; off <<= 1)
                rs += __shfl_xor(rs, off, 32);
            const float a = __expf(mrow[r] - mn);
            lrow[r] = lrow[r] * a + rs;
            mrow[r] = mn;
            alpha[r] = a;
            // P to LDS in 16x32 row-major (C layout: M = r + 8*half, N = lr)
            Pls[(r + 8 * half) * 32 + lr]      = (bf16)p0;
            Pls[(r + 8 * half) * 32 + 16 + lr] = (bf16)p1;
        }
        #pragma unroll
        for (int t = 0; t < 4; ++t)
            #pragma unroll
            for (int r = 0; r < 8; ++r) acc[t][r] *= alpha[r];

        asm volatile("s_wait_dscnt 0x0" ::: "memory");   // all lanes' P visible

        AF pa;  // P as A fragment (16x32, contraction = 32 keys)
        pa.h[0] = *(const v8bf*)(&Pls[lr * 32 + half * 16]);
        pa.h[1] = *(const v8bf*)(&Pls[lr * 32 + half * 16 + 8]);

        #pragma unroll
        for (int t = 0; t < 4; ++t) {
            const bf16* vrow = Vp + (size_t)(t * 16 + lr) * TSEQ + jp * 32;
            AF bv;
            bv.h[0] = *(const v8bf*)(vrow + half * 16);
            bv.h[1] = *(const v8bf*)(vrow + half * 16 + 8);
            acc[t] = wmma_bf16(pa, bv, acc[t]);
        }
    }

    // normalize and store ctx[b][t][h*64 + d] as bf16 for the output GEMM
    #pragma unroll
    for (int r = 0; r < 8; ++r) {
        const int trow = qt * 16 + r + 8 * half;
        const float inv = 1.0f / lrow[r];
        const size_t base = ((size_t)(b * TSEQ + trow)) * DIMX + h * HD;
        #pragma unroll
        for (int t = 0; t < 4; ++t)
            CTX[base + t * 16 + lr] = (bf16)(acc[t][r] * inv);
    }
}

// ------------------------------- launch ------------------------------------

extern "C" void kernel_launch(void* const* d_in, const int* in_sizes, int n_in,
                              void* d_out, int out_size, void* d_ws, size_t ws_size,
                              hipStream_t stream) {
    const float* x     = (const float*)d_in[0];   // (4,2048,768)
    const float* w_qkv = (const float*)d_in[1];   // (768,2304)
    const float* w_out = (const float*)d_in[2];   // (768,768)
    const float* b_out = (const float*)d_in[3];   // (768,)
    float* out = (float*)d_out;

    // workspace layout (bytes); total ~67.6 MB of bf16 intermediates
    char* ws = (char*)d_ws;
    const size_t SZ_X   = (size_t)MROWS * DIMX * 2;   // 12.58 MB
    const size_t SZ_WQ  = (size_t)NQKV * DIMX * 2;    //  3.54 MB
    const size_t SZ_WO  = (size_t)DIMX * DIMX * 2;    //  1.18 MB
    const size_t SZ_BHD = (size_t)MROWS * HD * HEADS * 2; // 12.58 MB

    bf16* Xb    = (bf16*)(ws);
    bf16* WqT   = (bf16*)(ws + SZ_X);
    bf16* WoT   = (bf16*)(ws + SZ_X + SZ_WQ);
    bf16* Qb    = (bf16*)(ws + SZ_X + SZ_WQ + SZ_WO);
    bf16* Kb    = (bf16*)(ws + SZ_X + SZ_WQ + SZ_WO + SZ_BHD);
    bf16* Vtb   = (bf16*)(ws + SZ_X + SZ_WQ + SZ_WO + 2 * SZ_BHD);
    bf16* Ctx   = (bf16*)(ws + SZ_X + SZ_WQ + SZ_WO + 3 * SZ_BHD);

    const int nx = MROWS * DIMX;
    cast_bf16_kernel<<<(nx + 255) / 256, 256, 0, stream>>>(x, Xb, nx);
    transpose_cast_kernel<<<(DIMX * NQKV + 255) / 256, 256, 0, stream>>>(w_qkv, WqT, DIMX, NQKV);
    transpose_cast_kernel<<<(DIMX * DIMX + 255) / 256, 256, 0, stream>>>(w_out, WoT, DIMX, DIMX);

    qkv_gemm_kernel<<<dim3(NQKV / 256, MROWS / 32), 256, 0, stream>>>(Xb, WqT, Qb, Kb, Vtb);
    attn_kernel<<<dim3(TSEQ / 16, HEADS, BATCH), 32, 0, stream>>>(Qb, Kb, Vtb, Ctx);
    out_gemm_kernel<<<dim3(DIMX / 256, MROWS / 32), 256, 0, stream>>>(Ctx, WoT, b_out, out);
}
